// ConsistGuidedEnhancer_40149354283873
// MI455X (gfx1250) — matrix operand order repaired and checked by reading, not compiled, not run on hardware
//
#include <hip/hip_runtime.h>
#include <cstdint>
#include <cstddef>

// ---------------------------------------------------------------------------
// ConsistGuidedEnhancer for MI455X (gfx1250, wave32, WMMA).
// All heavy matmuls go through v_wmma_f32_16x16x32_bf16 (bf16 in, f32 acc).
// ---------------------------------------------------------------------------

#define DEVINL __device__ __forceinline__

typedef __attribute__((ext_vector_type(16))) __bf16 bf16x16;
typedef __attribute__((ext_vector_type(8)))  float  f32x8;

DEVINL unsigned short f2bf(float f) {
  union { float f; unsigned int u; } v; v.f = f;
  unsigned int u = v.u;
  u += 0x7FFFu + ((u >> 16) & 1u);          // round-to-nearest-even
  return (unsigned short)(u >> 16);
}
DEVINL float gelu_f(float x) { return 0.5f * x * (1.0f + erff(x * 0.70710678118654752f)); }

// ---------------------------------------------------------------------------
// WMMA GEMM: C[M,N] = epilogue( A[M,K](bf16) @ B[K,N](bf16) )
// Block: 128x128 tile, 256 threads (8 waves). Wave w -> 32x64 subtile,
// 2x4 fragments of 16x16, K stepped by 32 (one v_wmma_f32_16x16x32_bf16 each).
// Requires: K % 32 == 0, N % 128 == 0 (true for every call site); M guarded.
// Epilogue (fully unrolled so accumulators never leave VGPRs): optional GELU,
// optional per-column gamma, optional residual add, optional f32 store,
// optional bf16 store (feeds the next GEMM).
// ---------------------------------------------------------------------------
__global__ __launch_bounds__(256)
void k_gemm(const unsigned short* __restrict__ A, long long strideA, int lda,
            const unsigned short* __restrict__ Bw, int ldb,
            float* __restrict__ C, long long strideC, int ldc,
            const float* __restrict__ Res, long long strideR, int ldr,
            const float* __restrict__ gamma,
            unsigned short* __restrict__ Obf, long long strideO, int ldo,
            int M, int N, int K, int gelu_flag)
{
  (void)N;
  __shared__ unsigned short sA[128 * 36];   // [m][k], padded stride 36
  __shared__ unsigned short sB[128 * 36];   // [n][k], padded stride 36 (B^T)

  const int tid    = threadIdx.x;
  const int lane   = tid & 31;
  const int laneLo = lane & 15;
  const int laneHi = lane >> 4;
  const int wave   = tid >> 5;
  const int mW     = (wave >> 1) * 32;
  const int nW     = (wave & 1) * 64;
  const int mTile  = blockIdx.y * 128;
  const int nTile  = blockIdx.x * 128;
  const int bat    = blockIdx.z;

  const unsigned short* Ab = A + (size_t)bat * (size_t)strideA;

  f32x8 acc[2][4];
#pragma unroll
  for (int i = 0; i < 2; i++)
#pragma unroll
    for (int j = 0; j < 4; j++)
#pragma unroll
      for (int r = 0; r < 8; r++) acc[i][j][r] = 0.0f;

  const int arow = tid >> 1, ahalf = (tid & 1) * 16;   // A: 2 threads/row, 16 bf16 each
  const int bk = tid >> 3, bn = (tid & 7) * 16;        // B: 16 n-values per thread

  for (int k0 = 0; k0 < K; k0 += 32) {
    __syncthreads();
    { // stage A tile: 128 x 32 bf16, 128-bit global loads
      int rowG = mTile + arow;
      uint4 d0 = {0, 0, 0, 0}, d1 = {0, 0, 0, 0};
      if (rowG < M) {
        const uint4* p = (const uint4*)(Ab + (size_t)rowG * lda + k0 + ahalf);
        d0 = p[0]; d1 = p[1];
      }
      unsigned int* ps = (unsigned int*)(sA + arow * 36 + ahalf);
      ps[0] = d0.x; ps[1] = d0.y; ps[2] = d0.z; ps[3] = d0.w;
      ps[4] = d1.x; ps[5] = d1.y; ps[6] = d1.z; ps[7] = d1.w;
    }
    { // stage B tile transposed: sB[n][k] <- B[k0+k][nTile+n]
      const uint4* p = (const uint4*)(Bw + (size_t)(k0 + bk) * ldb + nTile + bn);
      union { uint4 q[2]; unsigned short s[16]; } tv;
      tv.q[0] = p[0]; tv.q[1] = p[1];
#pragma unroll
      for (int j = 0; j < 16; j++) sB[(bn + j) * 36 + bk] = tv.s[j];
    }
    __syncthreads();

    union FR { unsigned int u[8]; bf16x16 v; };
    FR a0, a1, bf[4];
    // A fragment per ISA layout: lanes0-15 K{0..7,16..23}, lanes16-31 K{8..15,24..31}
#pragma unroll
    for (int v = 0; v < 8; v++) {
      int kIdx = (v & 3) * 2 + (v >> 2) * 16 + laneHi * 8;
      a0.u[v] = *(const unsigned int*)(sA + (mW + laneLo) * 36 + kIdx);
      a1.u[v] = *(const unsigned int*)(sA + (mW + 16 + laneLo) * 36 + kIdx);
    }
    // B fragment: same K layout as A (lanes0-15 K{0..7,16..23}, lanes16-31 K{8..15,24..31}), N = lane%16
#pragma unroll
    for (int ni = 0; ni < 4; ni++)
#pragma unroll
      for (int v = 0; v < 8; v++)
        bf[ni].u[v] = *(const unsigned int*)(sB + (nW + ni * 16 + laneLo) * 36 + (v & 3) * 2 + (v >> 2) * 16 + laneHi * 8);

#pragma unroll
    for (int ni = 0; ni < 4; ni++) {
      acc[0][ni] = __builtin_amdgcn_wmma_f32_16x16x32_bf16(false, a0.v, false, bf[ni].v,
                                                           (short)0, acc[0][ni], false, false);
      acc[1][ni] = __builtin_amdgcn_wmma_f32_16x16x32_bf16(false, a1.v, false, bf[ni].v,
                                                           (short)0, acc[1][ni], false, false);
    }
  }

  // epilogue: C/D layout -> VGPR r: lanes0-15 (M=r,N=lane), lanes16-31 (M=8+r,N=lane-16)
  // Fully unrolled: acc stays in VGPRs (no scratch spill).
  const float* ResB = Res ? Res + (size_t)bat * strideR : nullptr;
  float* CB = C ? C + (size_t)bat * strideC : nullptr;
  unsigned short* OB = Obf ? Obf + (size_t)bat * strideO : nullptr;
#pragma unroll
  for (int mi = 0; mi < 2; mi++)
#pragma unroll
    for (int ni = 0; ni < 4; ni++) {
#pragma unroll
      for (int r = 0; r < 8; r++) {
        int row = mTile + mW + mi * 16 + laneHi * 8 + r;
        if (row >= M) continue;
        int col = nTile + nW + ni * 16 + laneLo;
        float v = acc[mi][ni][r];
        if (gelu_flag) v = gelu_f(v);
        if (gamma) v *= gamma[col];
        if (ResB) v += ResB[(size_t)row * ldr + col];
        if (CB)   CB[(size_t)row * ldc + col] = v;
        if (OB)   OB[(size_t)row * ldo + col] = f2bf(v);
      }
    }
}

// ---------------------------------------------------------------------------
// LayerNorm over last dim, emits bf16 (A operand for the next GEMM).
// ---------------------------------------------------------------------------
__global__ void k_ln(const float* __restrict__ X, const float* __restrict__ g,
                     const float* __restrict__ bta, unsigned short* __restrict__ Y, int width)
{
  __shared__ float s1[256], s2[256];
  const int row = blockIdx.x, tid = threadIdx.x;
  const float* xr = X + (size_t)row * width;
  float s = 0.f, q = 0.f;
  for (int c = tid; c < width; c += 256) { float v = xr[c]; s += v; q += v * v; }
  s1[tid] = s; s2[tid] = q; __syncthreads();
  for (int st = 128; st; st >>= 1) {
    if (tid < st) { s1[tid] += s1[tid + st]; s2[tid] += s2[tid + st]; }
    __syncthreads();
  }
  float mean = s1[0] / width;
  float var  = s2[0] / width - mean * mean;
  float rstd = rsqrtf(var + 1e-5f);
  unsigned short* yr = Y + (size_t)row * width;
  for (int c = tid; c < width; c += 256)
    yr[c] = f2bf((xr[c] - mean) * rstd * g[c] + bta[c]);
}

// f32 -> bf16 (weights)
__global__ void k_cvt(const float* __restrict__ W, unsigned short* __restrict__ O, int n)
{
  int i = blockIdx.x * 256 + threadIdx.x;
  if (i < n) O[i] = f2bf(W[i]);
}

// group-transposed weight convert: W[(g*OCg+oc)*IC+ic] -> O[((g*IC+ic)*OCg)+oc]
__global__ void k_cvt_t(const float* __restrict__ W, unsigned short* __restrict__ O,
                        int G, int OCg, int IC)
{
  int i = blockIdx.x * 256 + threadIdx.x;
  int total = G * OCg * IC;
  if (i >= total) return;
  int oc = i % OCg;
  int ic = (i / OCg) % IC;
  int g  = i / (OCg * IC);
  O[i] = f2bf(W[((size_t)(g * OCg + oc)) * IC + ic]);
}

// ---------------------------------------------------------------------------
// 5x5 avg pool (separable, zero pad, /25) on token-major [B][HW][256] maps.
// ---------------------------------------------------------------------------
__global__ void k_pool_h(const float* __restrict__ I, float* __restrict__ O)
{
  int i = blockIdx.x * 256 + threadIdx.x;
  if (i >= 4 * 4096 * 256) return;
  int p = (i >> 8) & 4095;
  int ww = p & 63;
  float s = 0.f;
#pragma unroll
  for (int dx = -2; dx <= 2; dx++) {
    int w2 = ww + dx;
    if (w2 >= 0 && w2 < 64) s += I[i + dx * 256];
  }
  O[i] = s;
}
__global__ void k_pool_v(const float* __restrict__ I, float* __restrict__ O)
{
  int i = blockIdx.x * 256 + threadIdx.x;
  if (i >= 4 * 4096 * 256) return;
  int p = (i >> 8) & 4095;
  int hh = p >> 6;
  float s = 0.f;
#pragma unroll
  for (int dy = -2; dy <= 2; dy++) {
    int h2 = hh + dy;
    if (h2 >= 0 && h2 < 64) s += I[i + dy * 64 * 256];
  }
  O[i] = s * 0.04f;   // 1/25 (count_include_pad)
}
// gf_raw[b][p][s*256+c] = x2*y2
__global__ void k_prod(const float* __restrict__ X, const float* __restrict__ Y,
                       float* __restrict__ GF, int sidx)
{
  int i = blockIdx.x * 256 + threadIdx.x;
  if (i >= 4 * 4096 * 256) return;
  int c = i & 255;
  int p = (i >> 8) & 4095;
  int b = i >> 20;
  GF[((size_t)(b * 4096 + p)) * 2304 + sidx * 256 + c] = X[i] * Y[i];
}

// xy = concat(x, y) along tokens
__global__ void k_concat(const float* __restrict__ X, const float* __restrict__ Y,
                         float* __restrict__ XY)
{
  int i = blockIdx.x * 256 + threadIdx.x;
  if (i >= 4 * 8192 * 256) return;
  int c = i & 255;
  int r = (i >> 8) & 8191;
  int b = i >> 21;
  XY[i] = (r < 4096) ? X[(size_t)(b * 4096 + r) * 256 + c]
                     : Y[(size_t)(b * 4096 + r - 4096) * 256 + c];
}

// per-head l2 normalize (32-dim) in place
__global__ void k_l2n(float* __restrict__ P, int nRows, int stride, int off, int heads)
{
  int i = blockIdx.x * 256 + threadIdx.x;
  if (i >= nRows * heads) return;
  int row = i / heads, h = i % heads;
  float* p = P + (size_t)row * stride + off + h * 32;
  float s = 0.f;
#pragma unroll
  for (int d = 0; d < 32; d++) s += p[d] * p[d];
  float inv = 1.f / fmaxf(sqrtf(s), 1e-12f);
#pragma unroll
  for (int d = 0; d < 32; d++) p[d] *= inv;
}

// attn0 scores: SC[b][n][k][l] = <q0[k,head n], k0[b,l,head n]>
__global__ void k_scores0(const float* __restrict__ Q, const float* __restrict__ KV,
                          float* __restrict__ SC)
{
  int i = blockIdx.x * 256 + threadIdx.x;
  if (i >= 4 * 8 * 64 * 4096) return;
  int l  = i & 4095;
  int kq = (i >> 12) & 63;
  int n  = (i >> 18) & 7;
  int b  = i >> 21;
  const float* q  = Q + kq * 256 + n * 32;
  const float* kk = KV + ((size_t)(b * 4096 + l)) * 512 + n * 32;
  float s = 0.f;
#pragma unroll
  for (int d = 0; d < 32; d++) s += q[d] * kk[d];
  SC[i] = s;
}

// row softmax (in place)
__global__ void k_softmax(float* __restrict__ SC, int rowLen)
{
  __shared__ float sh[256];
  size_t base = (size_t)blockIdx.x * rowLen;
  int tid = threadIdx.x;
  float mx = -3.4e38f;
  for (int c = tid; c < rowLen; c += 256) mx = fmaxf(mx, SC[base + c]);
  sh[tid] = mx; __syncthreads();
  for (int st = 128; st; st >>= 1) { if (tid < st) sh[tid] = fmaxf(sh[tid], sh[tid + st]); __syncthreads(); }
  mx = sh[0]; __syncthreads();
  float sum = 0.f;
  for (int c = tid; c < rowLen; c += 256) { float e = expf(SC[base + c] - mx); SC[base + c] = e; sum += e; }
  sh[tid] = sum; __syncthreads();
  for (int st = 128; st; st >>= 1) { if (tid < st) sh[tid] += sh[tid + st]; __syncthreads(); }
  float inv = 1.f / sh[0];
  for (int c = tid; c < rowLen; c += 256) SC[base + c] *= inv;
}

// attn0 output -> token-major bf16 A for qkv0_w GEMM
__global__ void k_qkv0(const float* __restrict__ SC, const float* __restrict__ KV,
                       unsigned short* __restrict__ Aq)
{
  int i = blockIdx.x * 256 + threadIdx.x;
  if (i >= 4 * 8 * 64 * 32) return;
  int d  = i & 31;
  int kq = (i >> 5) & 63;
  int n  = (i >> 11) & 7;
  int b  = i >> 14;
  const float* row = SC + (size_t)((b * 8 + n) * 64 + kq) * 4096;
  const float* v   = KV + 256 + n * 32 + d;
  float s = 0.f;
  for (int l = 0; l < 4096; l++) s += row[l] * v[(size_t)(b * 4096 + l) * 512];
  Aq[(size_t)(b * 64 + kq) * 256 + n * 32 + d] = f2bf(s);
}

// attn1: per (b,head) block, K/V (64x32) staged in LDS; one token per thread.
__global__ __launch_bounds__(256)
void k_attn1(const float* __restrict__ Q, const float* __restrict__ KV,
             unsigned short* __restrict__ Ao)
{
  __shared__ float kS[64 * 32], vS[64 * 32];
  int bn = blockIdx.x;
  int b = bn >> 3, n = bn & 7;
  int tid = threadIdx.x;
  for (int i = tid; i < 2048; i += 256) {
    int kk = i >> 5, d = i & 31;
    kS[i] = KV[(size_t)(b * 64 + kk) * 512 + n * 32 + d];
    vS[i] = KV[(size_t)(b * 64 + kk) * 512 + 256 + n * 32 + d];
  }
  __syncthreads();
  int l = blockIdx.y * 256 + tid;
  const float* q = Q + (size_t)(b * 8192 + l) * 256 + n * 32;
  float qv[32];
#pragma unroll
  for (int d = 0; d < 32; d++) qv[d] = q[d];
  float s[64];
  float mx = -3.4e38f;
  for (int kk = 0; kk < 64; kk++) {
    float a = 0.f;
#pragma unroll
    for (int d = 0; d < 32; d++) a += qv[d] * kS[kk * 32 + d];
    s[kk] = a; mx = fmaxf(mx, a);
  }
  float sum = 0.f;
  for (int kk = 0; kk < 64; kk++) { float e = expf(s[kk] - mx); s[kk] = e; sum += e; }
  float inv = 1.f / sum;
  unsigned short* o = Ao + (size_t)(b * 8192 + l) * 256 + n * 32;
  for (int d = 0; d < 32; d++) {
    float a = 0.f;
    for (int kk = 0; kk < 64; kk++) a += s[kk] * vS[kk * 32 + d];
    o[d] = f2bf(a * inv);
  }
}

// depthwise 3x3 (SAME, zero pad) + GELU, emits bf16 for the fc2 GEMM
__global__ void k_dw(const float* __restrict__ T, const float* __restrict__ Wd,
                     unsigned short* __restrict__ O)
{
  int i = blockIdx.x * 256 + threadIdx.x;
  if (i >= 4 * 2 * 4096 * 1024) return;
  int cc = i & 1023;
  int p  = (i >> 10) & 4095;
  int bg = i >> 22;
  int g  = bg & 1;
  int hh = p >> 6, ww = p & 63;
  const float* wp = Wd + (size_t)(g * 1024 + cc) * 9;
  float s = 0.f;
#pragma unroll
  for (int dy = -1; dy <= 1; dy++) {
    int h2 = hh + dy; if (h2 < 0 || h2 > 63) continue;
#pragma unroll
    for (int dx = -1; dx <= 1; dx++) {
      int w2 = ww + dx; if (w2 < 0 || w2 > 63) continue;
      s += wp[(dy + 1) * 3 + dx + 1] * T[((size_t)bg * 4096 + h2 * 64 + w2) * 1024 + cc];
    }
  }
  O[i] = f2bf(gelu_f(s));
}

// guide map: channel mean, then per-batch min-max normalize
__global__ void k_gmmean(const float* __restrict__ G, float* __restrict__ M)
{
  int i = blockIdx.x * 256 + threadIdx.x;
  if (i >= 16384) return;
  const float* r = G + (size_t)i * 256;
  float s = 0.f;
  for (int c = 0; c < 256; c++) s += r[c];
  M[i] = s * (1.0f / 256.0f);
}
__global__ void k_gmnorm(const float* __restrict__ G, float* __restrict__ O)
{
  __shared__ float smn[256], smx[256];
  int b = blockIdx.x, tid = threadIdx.x;
  float mn = 3.4e38f, mx = -3.4e38f;
  for (int p = tid; p < 4096; p += 256) {
    float v = G[b * 4096 + p];
    mn = fminf(mn, v); mx = fmaxf(mx, v);
  }
  smn[tid] = mn; smx[tid] = mx; __syncthreads();
  for (int st = 128; st; st >>= 1) {
    if (tid < st) { smn[tid] = fminf(smn[tid], smn[tid + st]); smx[tid] = fmaxf(smx[tid], smx[tid + st]); }
    __syncthreads();
  }
  float lo = smn[0];
  float inv = 1.f / (smx[0] - lo);
  for (int p = tid; p < 4096; p += 256) O[b * 4096 + p] = (G[b * 4096 + p] - lo) * inv;
}

// ---------------------------------------------------------------------------
// host
// ---------------------------------------------------------------------------
extern "C" void kernel_launch(void* const* d_in, const int* in_sizes, int n_in,
                              void* d_out, int out_size, void* d_ws, size_t ws_size,
                              hipStream_t stream)
{
  (void)in_sizes; (void)n_in; (void)out_size; (void)ws_size;
  const float* x       = (const float*)d_in[0];
  const float* y       = (const float*)d_in[1];
  const float* p2d0_g  = (const float*)d_in[2];
  const float* p2d0_b  = (const float*)d_in[3];
  const float* p2d0_w  = (const float*)d_in[4];
  const float* p2d1_g  = (const float*)d_in[5];
  const float* p2d1_b  = (const float*)d_in[6];
  const float* p2d1_w  = (const float*)d_in[7];
  const float* merge_g = (const float*)d_in[8];
  const float* merge_b = (const float*)d_in[9];
  const float* merge_w1= (const float*)d_in[10];
  const float* merge_w2= (const float*)d_in[11];
  const float* proto   = (const float*)d_in[12];
  const float* q0_g    = (const float*)d_in[13];
  const float* q0_b    = (const float*)d_in[14];
  const float* q0_w    = (const float*)d_in[15];
  const float* kv0_g   = (const float*)d_in[16];
  const float* kv0_b   = (const float*)d_in[17];
  const float* kv0_w   = (const float*)d_in[18];
  const float* qkv0_w  = (const float*)d_in[19];
  const float* f0_g    = (const float*)d_in[20];
  const float* f0_b    = (const float*)d_in[21];
  const float* f0_w1   = (const float*)d_in[22];
  const float* f0_w2   = (const float*)d_in[23];
  const float* q1_g    = (const float*)d_in[24];
  const float* q1_b    = (const float*)d_in[25];
  const float* q1_w    = (const float*)d_in[26];
  const float* kv1_g   = (const float*)d_in[27];
  const float* kv1_b   = (const float*)d_in[28];
  const float* kv1_w   = (const float*)d_in[29];
  const float* qkv1_w  = (const float*)d_in[30];
  const float* f1_g    = (const float*)d_in[31];
  const float* f1_b    = (const float*)d_in[32];
  const float* f1_fc1  = (const float*)d_in[33];
  const float* f1_dw   = (const float*)d_in[34];
  const float* f1_fc2  = (const float*)d_in[35];
  const float* gammaP  = (const float*)d_in[36];
  float* out = (float*)d_out;

  // ---- workspace bump allocator with region overlays ----
  char* W8 = (char*)d_ws;
  size_t off = 0;
  auto alloc = [&](size_t bytes) { size_t r = off; off += (bytes + 255) & ~(size_t)255; return r; };
  auto Fp = [&](size_t o) { return (float*)(W8 + o); };
  auto Hp = [&](size_t o) { return (unsigned short*)(W8 + o); };

  const size_t o_wp2d0 = alloc(65536 * 2);
  const size_t o_wp2d1 = alloc(65536 * 2);
  const size_t o_wmw1  = alloc((size_t)2304 * 256 * 2);
  const size_t o_wmw2  = alloc(65536 * 2);
  const size_t o_wq0   = alloc(65536 * 2);
  const size_t o_wkv0  = alloc(131072 * 2);
  const size_t o_wqkv0 = alloc(65536 * 2);
  const size_t o_wf0w1 = alloc(262144 * 2);
  const size_t o_wf0w2 = alloc(262144 * 2);
  const size_t o_wq1   = alloc(65536 * 2);
  const size_t o_wkv1  = alloc(131072 * 2);
  const size_t o_wqkv1 = alloc(65536 * 2);
  const size_t o_w1t   = alloc((size_t)524288 * 2);
  const size_t o_w2t   = alloc((size_t)524288 * 2);
  const size_t o_R2 = alloc(50331648);   // x2|y2|tmp  -> later q1|Ao
  const size_t o_R1 = alloc(150994944);  // gf_raw     -> later fc1_out
  const size_t o_R3 = alloc(75497472);   // ln_big     -> later xy
  const size_t o_R4 = alloc(67108864);   // kv0|sc0    -> later t2
  const size_t o_lnmed  = alloc(16777216);
  const size_t o_lntiny = alloc(131072);
  const size_t o_h1  = alloc(8388608);
  const size_t o_gfm = alloc(16777216);
  const size_t o_c0  = alloc(65536);
  const size_t o_c1  = alloc(262144);
  const size_t o_f0h = alloc(524288);
  const size_t o_c2  = alloc(262144);
  const size_t o_kv1 = alloc(524288);
  const size_t o_aq0 = alloc(131072);
  const size_t o_gmr = alloc(65536);

  float* x2   = Fp(o_R2);
  float* y2   = Fp(o_R2 + 16777216);
  float* tmpb = Fp(o_R2 + 33554432);
  float* q1b  = Fp(o_R2);                       // overlay (after pyramid)
  unsigned short* Ao = Hp(o_R2 + 33554432);     // overlay
  float* gfraw = Fp(o_R1);
  float* fc1o  = Fp(o_R1);                      // overlay (after gf LN)
  unsigned short* lnbig = Hp(o_R3);
  float* xy = Fp(o_R3);                         // overlay (after merge GEMM1)
  float* kv0 = Fp(o_R4);
  float* sc0 = Fp(o_R4 + 33554432);
  unsigned short* t2 = Hp(o_R4);                // overlay (after attn0)

  auto nb = [](long long n) { return dim3((unsigned)((n + 255) / 256)); };

  auto gemm = [&](const unsigned short* A, long long sA, int lda,
                  const unsigned short* B, int ldb,
                  float* C, long long sC, int ldc,
                  const float* R, long long sR, int ldr,
                  const float* gm, unsigned short* O, long long sO, int ldo,
                  int M, int N, int K, int gelu, int batch) {
    dim3 g((unsigned)(N / 128), (unsigned)((M + 127) / 128), (unsigned)batch);
    k_gemm<<<g, 256, 0, stream>>>(A, sA, lda, B, ldb, C, sC, ldc, R, sR, ldr, gm, O, sO, ldo, M, N, K, gelu);
  };

  // ---- weight conversions to bf16 ----
  k_cvt<<<nb(65536), 256, 0, stream>>>(p2d0_w, Hp(o_wp2d0), 65536);
  k_cvt<<<nb(65536), 256, 0, stream>>>(p2d1_w, Hp(o_wp2d1), 65536);
  k_cvt<<<nb(589824), 256, 0, stream>>>(merge_w1, Hp(o_wmw1), 589824);
  k_cvt<<<nb(65536), 256, 0, stream>>>(merge_w2, Hp(o_wmw2), 65536);
  k_cvt<<<nb(65536), 256, 0, stream>>>(q0_w, Hp(o_wq0), 65536);
  k_cvt<<<nb(131072), 256, 0, stream>>>(kv0_w, Hp(o_wkv0), 131072);
  k_cvt<<<nb(65536), 256, 0, stream>>>(qkv0_w, Hp(o_wqkv0), 65536);
  k_cvt<<<nb(262144), 256, 0, stream>>>(f0_w1, Hp(o_wf0w1), 262144);
  k_cvt<<<nb(262144), 256, 0, stream>>>(f0_w2, Hp(o_wf0w2), 262144);
  k_cvt<<<nb(65536), 256, 0, stream>>>(q1_w, Hp(o_wq1), 65536);
  k_cvt<<<nb(131072), 256, 0, stream>>>(kv1_w, Hp(o_wkv1), 131072);
  k_cvt<<<nb(65536), 256, 0, stream>>>(qkv1_w, Hp(o_wqkv1), 65536);
  k_cvt_t<<<nb(524288), 256, 0, stream>>>(f1_fc1, Hp(o_w1t), 2, 1024, 256);   // -> [g][256][1024]
  k_cvt_t<<<nb(524288), 256, 0, stream>>>(f1_fc2, Hp(o_w2t), 2, 256, 1024);   // -> [g][1024][256]

  // ---- guide features: x2 = LN(x)@p2d0_w, y2 = LN(y)@p2d1_w ----
  k_ln<<<16384, 256, 0, stream>>>(x, p2d0_g, p2d0_b, Hp(o_lnmed), 256);
  gemm(Hp(o_lnmed), 0, 256, Hp(o_wp2d0), 256, x2, 0, 256,
       nullptr, 0, 0, nullptr, nullptr, 0, 0, 16384, 256, 256, 0, 1);
  k_ln<<<16384, 256, 0, stream>>>(y, p2d1_g, p2d1_b, Hp(o_lnmed), 256);
  gemm(Hp(o_lnmed), 0, 256, Hp(o_wp2d1), 256, y2, 0, 256,
       nullptr, 0, 0, nullptr, nullptr, 0, 0, 16384, 256, 256, 0, 1);

  // ---- multi-scale pooled products -> gf_raw [b][hw][2304] ----
  for (int s = 0; s < 9; s++) {
    k_prod<<<nb(4194304), 256, 0, stream>>>(x2, y2, gfraw, s);
    if (s < 8) {
      k_pool_h<<<nb(4194304), 256, 0, stream>>>(x2, tmpb);
      k_pool_v<<<nb(4194304), 256, 0, stream>>>(tmpb, x2);
      k_pool_h<<<nb(4194304), 256, 0, stream>>>(y2, tmpb);
      k_pool_v<<<nb(4194304), 256, 0, stream>>>(tmpb, y2);
    }
  }

  // ---- merge MLP: gf = GELU(LN(gf_raw)@w1)@w2 ----
  k_ln<<<16384, 256, 0, stream>>>(gfraw, merge_g, merge_b, lnbig, 2304);
  gemm(lnbig, 0, 2304, Hp(o_wmw1), 256, nullptr, 0, 0,
       nullptr, 0, 0, nullptr, Hp(o_h1), 0, 256, 16384, 256, 2304, 1, 1);
  gemm(Hp(o_h1), 0, 256, Hp(o_wmw2), 256, Fp(o_gfm), 0, 256,
       nullptr, 0, 0, nullptr, nullptr, 0, 0, 16384, 256, 256, 0, 1);

  // xy = concat(x, y)   (ln_big region is dead now)
  k_concat<<<nb(8388608), 256, 0, stream>>>(x, y, xy);

  // ---- attn0: prototypes x guide ----
  k_ln<<<16384, 256, 0, stream>>>(Fp(o_gfm), kv0_g, kv0_b, Hp(o_lnmed), 256);
  gemm(Hp(o_lnmed), 0, 256, Hp(o_wkv0), 512, kv0, 0, 512,
       nullptr, 0, 0, nullptr, nullptr, 0, 0, 16384, 512, 256, 0, 1);
  k_ln<<<64, 256, 0, stream>>>(proto, q0_g, q0_b, Hp(o_lntiny), 256);
  gemm(Hp(o_lntiny), 0, 256, Hp(o_wq0), 256, Fp(o_c0), 0, 256,
       nullptr, 0, 0, nullptr, nullptr, 0, 0, 64, 256, 256, 0, 1);
  k_l2n<<<nb(512), 256, 0, stream>>>(Fp(o_c0), 64, 256, 0, 8);
  k_l2n<<<nb(131072), 256, 0, stream>>>(kv0, 16384, 512, 0, 8);
  k_scores0<<<nb(8388608), 256, 0, stream>>>(Fp(o_c0), kv0, sc0);
  k_softmax<<<2048, 256, 0, stream>>>(sc0, 4096);
  k_qkv0<<<nb(65536), 256, 0, stream>>>(sc0, kv0, Hp(o_aq0));
  gemm(Hp(o_aq0), 0, 256, Hp(o_wqkv0), 256, Fp(o_c1), 0, 256,
       nullptr, 0, 0, nullptr, nullptr, 0, 0, 256, 256, 256, 0, 1);

  // ---- f0 MLP on centers (no residual) ----
  k_ln<<<256, 256, 0, stream>>>(Fp(o_c1), f0_g, f0_b, Hp(o_lntiny), 256);
  gemm(Hp(o_lntiny), 0, 256, Hp(o_wf0w1), 1024, nullptr, 0, 0,
       nullptr, 0, 0, nullptr, Hp(o_f0h), 0, 1024, 256, 1024, 256, 1, 1);
  gemm(Hp(o_f0h), 0, 1024, Hp(o_wf0w2), 256, Fp(o_c2), 0, 256,
       nullptr, 0, 0, nullptr, nullptr, 0, 0, 256, 256, 1024, 0, 1);

  // ---- attn1: all tokens x prototypes ----
  k_ln<<<256, 256, 0, stream>>>(Fp(o_c2), kv1_g, kv1_b, Hp(o_lntiny), 256);
  gemm(Hp(o_lntiny), 0, 256, Hp(o_wkv1), 512, Fp(o_kv1), 0, 512,
       nullptr, 0, 0, nullptr, nullptr, 0, 0, 256, 512, 256, 0, 1);
  k_ln<<<32768, 256, 0, stream>>>(xy, q1_g, q1_b, Hp(o_lnmed), 256);
  gemm(Hp(o_lnmed), 0, 256, Hp(o_wq1), 256, q1b, 0, 256,
       nullptr, 0, 0, nullptr, nullptr, 0, 0, 32768, 256, 256, 0, 1);
  k_l2n<<<nb(262144), 256, 0, stream>>>(q1b, 32768, 256, 0, 8);
  k_l2n<<<nb(2048), 256, 0, stream>>>(Fp(o_kv1), 256, 512, 0, 8);
  k_attn1<<<dim3(32, 32), 256, 0, stream>>>(q1b, Fp(o_kv1), Ao);
  // xy += (o @ qkv1_w) * gamma      (in-place residual)
  gemm(Ao, 0, 256, Hp(o_wqkv1), 256, xy, 0, 256,
       xy, 0, 256, gammaP, nullptr, 0, 0, 32768, 256, 256, 0, 1);

  // ---- grouped conv FFN ----
  k_ln<<<32768, 256, 0, stream>>>(xy, f1_g, f1_b, Hp(o_lnmed), 256);
  for (int g = 0; g < 2; g++) {   // fc1: 1x1 conv as GEMM per (batch, group)
    gemm(Hp(o_lnmed) + (size_t)g * 4096 * 256, (long long)8192 * 256, 256,
         Hp(o_w1t) + (size_t)g * 256 * 1024, 1024,
         fc1o + (size_t)g * 4096 * 1024, (long long)8388608, 1024,
         nullptr, 0, 0, nullptr, nullptr, 0, 0, 4096, 1024, 256, 0, 4);
  }
  k_dw<<<nb(33554432), 256, 0, stream>>>(fc1o, f1_dw, t2);   // dw3x3 + GELU -> bf16
  for (int g = 0; g < 2; g++) {   // fc2 + residual, writes x_out / y_out directly
    gemm(t2 + (size_t)g * 4096 * 1024, (long long)8192 * 1024, 1024,
         Hp(o_w2t) + (size_t)g * 1024 * 256, 256,
         out + (size_t)g * 4194304, (long long)1048576, 256,
         xy + (size_t)g * 4096 * 256, (long long)8192 * 256, 256,
         nullptr, nullptr, 0, 0, 4096, 256, 1024, 0, 4);
  }

  // ---- guide map ----
  k_gmmean<<<nb(16384), 256, 0, stream>>>(Fp(o_gfm), Fp(o_gmr));
  k_gmnorm<<<4, 256, 0, stream>>>(Fp(o_gmr), out + 8388608);
}
